// EncoderLayer_43181601194322
// MI455X (gfx1250) — compile-verified, matrix operand order checked by reading, not run on hardware
//
#include <hip/hip_runtime.h>
#include <hip/hip_bf16.h>

// ---------------------------------------------------------------------------
// Problem constants
// ---------------------------------------------------------------------------
#define BB   16
#define SS   512
#define DD   1024
#define HH   8
#define DHALF 512            // D/2
#define DHEAD 64             // DHALF/H
#define DFFN 4096
#define MM   (BB * SS)       // 8192 rows

typedef __attribute__((ext_vector_type(16))) __bf16 v16bf;
typedef __attribute__((ext_vector_type(8)))  __bf16 v8bf;
typedef __attribute__((ext_vector_type(4)))  __bf16 v4bf;
typedef __attribute__((ext_vector_type(8)))  float  v8f;

// LDS tile geometry (bf16): rows padded to 40 elements (80 B) -> rows stay
// 16-B aligned and (20*idx) mod 64 banks is distinct for idx = 0..15
// (conflict-free ds_load_b128 fragment reads).
#define LROW 40
#define LA_BYTES (128 * LROW * 2)    // 10240 B per A buffer
#define LB_BYTES (64 * LROW * 2)     //  5120 B per B buffer

// ---------------------------------------------------------------------------
// CDNA5 async global->LDS copy (16 B per lane), tracked with ASYNCcnt.
// ---------------------------------------------------------------------------
__device__ __forceinline__ void async_copy_b128(unsigned int lds_off, const void* gptr) {
    asm volatile("global_load_async_to_lds_b128 %0, %1, off"
                 :: "v"(lds_off), "v"(gptr) : "memory");
}
__device__ __forceinline__ void wait_async0() {
    asm volatile("s_wait_asynccnt 0x0" ::: "memory");
}

// ---------------------------------------------------------------------------
// bf16 fragment straight from an LDS row: two aligned 16-B loads, no VALU.
// Lane l (half = l>>4) holds elements 0..7  = K[8*half + 0..7],
//                            elements 8..15 = K[16 + 8*half + 0..7]
// (CDNA5 ISA 7.12.2, 16-bit A-matrix 16x32; pre-transposed B uses same loader.)
// ---------------------------------------------------------------------------
__device__ __forceinline__ v16bf frag_from_row(const __bf16* rowp, int lane) {
    const __bf16* p = rowp + ((lane >> 4) << 3);
    v8bf lo = *(const v8bf*)(p);        // K chunk 0..7   (+8*half)
    v8bf hi = *(const v8bf*)(p + 16);   // K chunk 16..23 (+8*half)
    v16bf f;
    #pragma unroll
    for (int i = 0; i < 8; ++i) { f[i] = lo[i]; f[i + 8] = hi[i]; }
    return f;
}

// ---------------------------------------------------------------------------
// Batched WMMA GEMM: C[m,n] = sum_k A[m,k] * Bt[n,k]  (+ epilogue)
// A, Bt are bf16 in memory. 256 threads = 8 waves (4 over M x 2 over N);
// block tile 128x64; wave 32x32. Tiles double-buffered in LDS via async DMA
// issued one K-step ahead; per-thread DMA pointers advance by 64 B per step.
// EPI: 0 = +bias(optional); 1 = +bias then GELU;
//      2 = local logits:  v/8 + mask[b,n]*-1e9 + adjoin[b,m,n]
//      3 = global logits: relu(v)*rescale(dist[b,m,n])/8 + mask[b,n]*-1e9
// OB:  0 = store fp32; 1 = store bf16
// ---------------------------------------------------------------------------
template<int EPI, int OB>
__global__ __launch_bounds__(256)
void gemm_wmma_bf16(const __bf16* __restrict__ A,  int lda, long aOut, long aIn,
                    const __bf16* __restrict__ Bt, int ldb, long bOut, long bIn,
                    void* __restrict__ Cv,         int ldc, long cOut, long cIn,
                    const float* __restrict__ bias,
                    const float* __restrict__ p1,   // mask  (B,1,1,S)
                    const float* __restrict__ p2,   // adjoin / dist (B,1,S,S)
                    int K, int innerDiv, int Scols) {
    __shared__ __align__(16) __bf16 lA[2][128 * LROW];
    __shared__ __align__(16) __bf16 lB[2][64 * LROW];

    const int tid  = threadIdx.x;
    const int lane = tid & 31;
    const int wid  = tid >> 5;
    const int z  = blockIdx.z;
    const int zo = z / innerDiv;
    const int zi = z - zo * innerDiv;
    A  += (size_t)zo * aOut + (size_t)zi * aIn;
    Bt += (size_t)zo * bOut + (size_t)zi * bIn;
    const size_t cOff = (size_t)zo * cOut + (size_t)zi * cIn;

    const int mBlk = blockIdx.x * 128;
    const int nBlk = blockIdx.y * 64;
    const int m0w  = (wid & 3) * 32;     // wave's M offset within block tile
    const int n0w  = (wid >> 2) * 32;    // wave's N offset within block tile
    const int idx  = lane & 15;

    const unsigned int baseA = (unsigned int)(size_t)&lA[0][0];
    const unsigned int baseB = (unsigned int)(size_t)&lB[0][0];

    // --- per-thread DMA assignments: A tile 128x32 bf16 = 512 x 16B (2/thread),
    //     B tile 64x32 bf16 = 256 x 16B (1/thread). Pointers bumped 32 elem/step.
    const int tA1 = tid + 256;
    const int rA0 = tid >> 2, cA0 = (tid & 3) << 3;
    const int rA1 = tA1 >> 2, cA1 = (tA1 & 3) << 3;
    const int rB  = tid >> 2, cB  = (tid & 3) << 3;
    const __bf16* pa0 = A  + (size_t)(mBlk + rA0) * lda + cA0;
    const __bf16* pa1 = A  + (size_t)(mBlk + rA1) * lda + cA1;
    const __bf16* pb0 = Bt + (size_t)(nBlk + rB)  * ldb + cB;
    const unsigned oa0 = baseA + (unsigned)(rA0 * (LROW * 2) + ((tid & 3) << 4));
    const unsigned oa1 = baseA + (unsigned)(rA1 * (LROW * 2) + ((tA1 & 3) << 4));
    const unsigned ob0 = baseB + (unsigned)(rB  * (LROW * 2) + ((tid & 3) << 4));

    v8f acc[2][2] = {};

    // Prologue: stage first K-step into buffer 0.
    async_copy_b128(oa0, pa0);
    async_copy_b128(oa1, pa1);
    async_copy_b128(ob0, pb0);
    pa0 += 32; pa1 += 32; pb0 += 32;
    int cur = 0;

    for (int k0 = 0; k0 < K; k0 += 32) {
        wait_async0();           // own DMA for buf[cur] complete
        __syncthreads();         // everyone's DMA complete, prev reads done
        if (k0 + 32 < K) {       // prefetch next step into the other buffer
            const unsigned aAdd = (unsigned)(cur ^ 1) * LA_BYTES;
            const unsigned bAdd = (unsigned)(cur ^ 1) * LB_BYTES;
            async_copy_b128(oa0 + aAdd, pa0);
            async_copy_b128(oa1 + aAdd, pa1);
            async_copy_b128(ob0 + bAdd, pb0);
            pa0 += 32; pa1 += 32; pb0 += 32;
        }

        v16bf a0 = frag_from_row(&lA[cur][(m0w + idx)      * LROW], lane);
        v16bf a1 = frag_from_row(&lA[cur][(m0w + 16 + idx) * LROW], lane);
        v16bf b0 = frag_from_row(&lB[cur][(n0w + idx)      * LROW], lane);
        v16bf b1 = frag_from_row(&lB[cur][(n0w + 16 + idx) * LROW], lane);
        acc[0][0] = __builtin_amdgcn_wmma_f32_16x16x32_bf16(false, a0, false, b0, (short)0, acc[0][0], false, false);
        acc[0][1] = __builtin_amdgcn_wmma_f32_16x16x32_bf16(false, a0, false, b1, (short)0, acc[0][1], false, false);
        acc[1][0] = __builtin_amdgcn_wmma_f32_16x16x32_bf16(false, a1, false, b0, (short)0, acc[1][0], false, false);
        acc[1][1] = __builtin_amdgcn_wmma_f32_16x16x32_bf16(false, a1, false, b1, (short)0, acc[1][1], false, false);
        cur ^= 1;
    }

    float*  Cf = (float*)Cv;
    __bf16* Cb = (__bf16*)Cv;
    const int half = lane >> 4;
    #pragma unroll
    for (int i = 0; i < 2; ++i) {
        #pragma unroll
        for (int j = 0; j < 2; ++j) {
            const int n = nBlk + n0w + 16 * j + idx;
            #pragma unroll
            for (int g = 0; g < 8; ++g) {
                const int m = mBlk + m0w + 16 * i + g + 8 * half;
                float v = acc[i][j][g];
                if (EPI == 0) {
                    if (bias) v += bias[n];
                } else if (EPI == 1) {
                    v += bias[n];
                    v = 0.5f * v * (1.0f + erff(v * 0.70710678118654752f));
                } else if (EPI == 2) {
                    v = v * 0.125f
                        + p1[(size_t)zo * Scols + n] * -1.0e9f
                        + p2[((size_t)zo * Scols + m) * Scols + n];
                } else { // EPI == 3
                    float dst = p2[((size_t)zo * Scols + m) * Scols + n];
                    float rs  = 3.71828182845904523f / (1.0f + __expf(1.0f - dst));
                    v = fmaxf(v, 0.0f) * rs * 0.125f
                        + p1[(size_t)zo * Scols + n] * -1.0e9f;
                }
                const size_t o = cOff + (size_t)m * ldc + n;
                if (OB) Cb[o] = (__bf16)v; else Cf[o] = v;
            }
        }
    }
}

// ---------------------------------------------------------------------------
// Bulk fp32 -> bf16 conversion (4 elements per thread, vector stores)
// ---------------------------------------------------------------------------
__global__ __launch_bounds__(256)
void f32_to_bf16(const float4* __restrict__ s, __bf16* __restrict__ d) {
    const size_t i = (size_t)blockIdx.x * 256 + threadIdx.x;
    float4 v = s[i];
    v4bf o; o[0] = (__bf16)v.x; o[1] = (__bf16)v.y; o[2] = (__bf16)v.z; o[3] = (__bf16)v.w;
    *(v4bf*)(d + i * 4) = o;
}

// ---------------------------------------------------------------------------
// Batched LDS-tiled transpose, fp32 in -> bf16 out: dst[c, r] = (bf16)src[r, c]
// ---------------------------------------------------------------------------
__global__ void transpose_f32_bf16(const float* __restrict__ src, int lds_, long sOut, long sIn,
                                   __bf16* __restrict__ dst, int ldd, long dOut, long dIn,
                                   int R, int Ccols, int innerDiv) {
    __shared__ float tile[32][33];
    const int z  = blockIdx.z;
    const int zo = z / innerDiv;
    const int zi = z - zo * innerDiv;
    src += (size_t)zo * sOut + (size_t)zi * sIn;
    dst += (size_t)zo * dOut + (size_t)zi * dIn;
    const int r0 = blockIdx.y * 32, c0 = blockIdx.x * 32;
    const int tx = threadIdx.x, ty = threadIdx.y;
    #pragma unroll
    for (int i = ty; i < 32; i += 8) {
        int r = r0 + i, c = c0 + tx;
        tile[i][tx] = (r < R && c < Ccols) ? src[(size_t)r * lds_ + c] : 0.0f;
    }
    __syncthreads();
    #pragma unroll
    for (int i = ty; i < 32; i += 8) {
        int c = c0 + i, r = r0 + tx;
        if (c < Ccols && r < R) dst[(size_t)c * ldd + r] = (__bf16)tile[tx][i];
    }
}

// ---------------------------------------------------------------------------
// In-place softmax over rows of 512 + bf16 shadow copy for the AV GEMM
// ---------------------------------------------------------------------------
__global__ __launch_bounds__(128)
void softmax512(float* __restrict__ data, __bf16* __restrict__ shadow) {
    __shared__ float red[128];
    const size_t rbase = (size_t)blockIdx.x * 512;
    float* row = data + rbase;
    const int t = threadIdx.x;
    float v[4];
    float mx = -3.4e38f;
    #pragma unroll
    for (int i = 0; i < 4; ++i) { v[i] = row[t + 128 * i]; mx = fmaxf(mx, v[i]); }
    red[t] = mx; __syncthreads();
    for (int s = 64; s > 0; s >>= 1) { if (t < s) red[t] = fmaxf(red[t], red[t + s]); __syncthreads(); }
    mx = red[0]; __syncthreads();
    float sum = 0.0f;
    #pragma unroll
    for (int i = 0; i < 4; ++i) { v[i] = __expf(v[i] - mx); sum += v[i]; }
    red[t] = sum; __syncthreads();
    for (int s = 64; s > 0; s >>= 1) { if (t < s) red[t] += red[t + s]; __syncthreads(); }
    const float inv = 1.0f / red[0];
    #pragma unroll
    for (int i = 0; i < 4; ++i) {
        const float r = v[i] * inv;
        row[t + 128 * i] = r;
        shadow[rbase + t + 128 * i] = (__bf16)r;
    }
}

// ---------------------------------------------------------------------------
// out = LayerNorm(x + y) * g + b  (D = 1024), optional bf16 shadow of out
// ---------------------------------------------------------------------------
__global__ __launch_bounds__(256)
void add_ln1024(const float* __restrict__ x, const float* __restrict__ y,
                const float* __restrict__ g, const float* __restrict__ b,
                float* __restrict__ out, __bf16* __restrict__ shadow) {
    __shared__ float red[256];
    const size_t row = (size_t)blockIdx.x * 1024;
    const int t = threadIdx.x;
    float v[4];
    float s = 0.0f, s2 = 0.0f;
    #pragma unroll
    for (int i = 0; i < 4; ++i) {
        int c = t + 256 * i;
        v[i] = x[row + c] + y[row + c];
        s += v[i]; s2 += v[i] * v[i];
    }
    red[t] = s; __syncthreads();
    for (int k = 128; k > 0; k >>= 1) { if (t < k) red[t] += red[t + k]; __syncthreads(); }
    const float mean = red[0] * (1.0f / 1024.0f); __syncthreads();
    red[t] = s2; __syncthreads();
    for (int k = 128; k > 0; k >>= 1) { if (t < k) red[t] += red[t + k]; __syncthreads(); }
    const float var = red[0] * (1.0f / 1024.0f) - mean * mean;
    const float inv = rsqrtf(var + 1e-5f);
    #pragma unroll
    for (int i = 0; i < 4; ++i) {
        int c = t + 256 * i;
        const float r = (v[i] - mean) * inv * g[c] + b[c];
        out[row + c] = r;
        if (shadow) shadow[row + c] = (__bf16)r;
    }
}

// ---------------------------------------------------------------------------
// Host launcher
// ---------------------------------------------------------------------------
extern "C" void kernel_launch(void* const* d_in, const int* in_sizes, int n_in,
                              void* d_out, int out_size, void* d_ws, size_t ws_size,
                              hipStream_t stream) {
    (void)in_sizes; (void)n_in; (void)out_size; (void)ws_size;

    const float* x      = (const float*)d_in[0];
    const float* mask   = (const float*)d_in[2];
    const float* adjoin = (const float*)d_in[3];
    const float* dist   = (const float*)d_in[4];
    // wq1,bq1,wk1,bk1,wv1,bv1,wo1,bo1, wq2,bq2,wk2,bk2,wv2,bv2,wo2,bo2 at 5..20
    const float* W[8]; const float* Bv[8];
    for (int i = 0; i < 8; ++i) { W[i] = (const float*)d_in[5 + 2 * i]; Bv[i] = (const float*)d_in[6 + 2 * i]; }
    const float* wff1 = (const float*)d_in[21]; const float* bff1 = (const float*)d_in[22];
    const float* wff2 = (const float*)d_in[23]; const float* bff2 = (const float*)d_in[24];
    const float* ln1g = (const float*)d_in[25]; const float* ln1b = (const float*)d_in[26];
    const float* ln2g = (const float*)d_in[27]; const float* ln2b = (const float*)d_in[28];

    float* out = (float*)d_out;
    const size_t OUT2  = (size_t)MM * DD;                 // 8,388,608
    const size_t AWSZ  = (size_t)BB * HH * SS * SS;       // 33,554,432
    float* awL = out + OUT2;
    float* awG = out + OUT2 + AWSZ;

    // ---- workspace carving (byte-based, 256-B aligned chunks) ----
    char* wsb = (char*)d_ws;
    size_t off = 0;
    auto carveB = [&](size_t bytes) { char* p = wsb + off; off += (bytes + 255) & ~(size_t)255; return (void*)p; };
    __bf16* Wt[8];
    for (int i = 0; i < 8; ++i) Wt[i] = (__bf16*)carveB((size_t)DHALF * DHALF * 2);
    __bf16* wff1t = (__bf16*)carveB((size_t)DFFN * DD * 2);
    __bf16* wff2t = (__bf16*)carveB((size_t)DD * DFFN * 2);
    __bf16* xb    = (__bf16*)carveB((size_t)MM * DD * 2);
    __bf16* q1    = (__bf16*)carveB((size_t)MM * DHALF * 2);
    __bf16* k1    = (__bf16*)carveB((size_t)MM * DHALF * 2);
    __bf16* q2    = (__bf16*)carveB((size_t)MM * DHALF * 2);
    __bf16* k2    = (__bf16*)carveB((size_t)MM * DHALF * 2);
    float*  v1    = (float*) carveB((size_t)MM * DHALF * 4);
    float*  v2    = (float*) carveB((size_t)MM * DHALF * 4);
    __bf16* Vt1   = (__bf16*)carveB((size_t)BB * HH * DHEAD * SS * 2);
    __bf16* Vt2   = (__bf16*)carveB((size_t)BB * HH * DHEAD * SS * 2);
    __bf16* awB   = (__bf16*)carveB(2 * AWSZ * 2);        // bf16 shadow of both aw
    __bf16* ctx1  = (__bf16*)carveB((size_t)MM * DHALF * 2);
    __bf16* ctx2  = (__bf16*)carveB((size_t)MM * DHALF * 2);
    float*  attn  = (float*) carveB((size_t)MM * DD * 4);
    float*  out1  = (float*) carveB((size_t)MM * DD * 4);
    __bf16* out1b = (__bf16*)carveB((size_t)MM * DD * 2);
    __bf16* hidden= (__bf16*)carveB((size_t)MM * DFFN * 2);
    float*  ffnout= (float*) carveB((size_t)MM * DD * 4);
    __bf16* awLb = awB;
    __bf16* awGb = awB + AWSZ;

    dim3 tb(32, 8);

    // ---- 0) x -> bf16 copy (A operand of the QKV GEMMs) ----
    f32_to_bf16<<<(unsigned)(OUT2 / (256 * 4)), 256, 0, stream>>>((const float4*)x, xb);

    // ---- 1) Transpose all weights to bf16: Wt[n,k] = W[k,n] ----
    for (int i = 0; i < 8; ++i)
        transpose_f32_bf16<<<dim3(DHALF / 32, DHALF / 32, 1), tb, 0, stream>>>(
            W[i], DHALF, 0, 0, Wt[i], DHALF, 0, 0, DHALF, DHALF, 1);
    transpose_f32_bf16<<<dim3(DFFN / 32, DD / 32, 1), tb, 0, stream>>>(
        wff1, DFFN, 0, 0, wff1t, DD, 0, 0, DD, DFFN, 1);
    transpose_f32_bf16<<<dim3(DD / 32, DFFN / 32, 1), tb, 0, stream>>>(
        wff2, DD, 0, 0, wff2t, DFFN, 0, 0, DFFN, DD, 1);

    // ---- 2) QKV projections (M=8192, N=512, K=512); q,k -> bf16, v -> fp32 ----
    dim3 gProj(MM / 128, DHALF / 64, 1);
    gemm_wmma_bf16<0,1><<<gProj, 256, 0, stream>>>(xb,         DD, 0, 0, Wt[0], DHALF, 0, 0, q1, DHALF, 0, 0, Bv[0], nullptr, nullptr, DHALF, 1, SS);
    gemm_wmma_bf16<0,1><<<gProj, 256, 0, stream>>>(xb,         DD, 0, 0, Wt[1], DHALF, 0, 0, k1, DHALF, 0, 0, Bv[1], nullptr, nullptr, DHALF, 1, SS);
    gemm_wmma_bf16<0,0><<<gProj, 256, 0, stream>>>(xb,         DD, 0, 0, Wt[2], DHALF, 0, 0, v1, DHALF, 0, 0, Bv[2], nullptr, nullptr, DHALF, 1, SS);
    gemm_wmma_bf16<0,1><<<gProj, 256, 0, stream>>>(xb + DHALF, DD, 0, 0, Wt[4], DHALF, 0, 0, q2, DHALF, 0, 0, Bv[4], nullptr, nullptr, DHALF, 1, SS);
    gemm_wmma_bf16<0,1><<<gProj, 256, 0, stream>>>(xb + DHALF, DD, 0, 0, Wt[5], DHALF, 0, 0, k2, DHALF, 0, 0, Bv[5], nullptr, nullptr, DHALF, 1, SS);
    gemm_wmma_bf16<0,0><<<gProj, 256, 0, stream>>>(xb + DHALF, DD, 0, 0, Wt[6], DHALF, 0, 0, v2, DHALF, 0, 0, Bv[6], nullptr, nullptr, DHALF, 1, SS);

    // ---- 3) Attention logits (fp32 into d_out), batched over z = b*H + h ----
    const long qkOut = (long)SS * DHALF, qkIn = DHEAD;
    const long awOut = (long)HH * SS * SS, awIn = (long)SS * SS;
    dim3 gScore(SS / 128, SS / 64, BB * HH);
    gemm_wmma_bf16<2,0><<<gScore, 256, 0, stream>>>(
        q1, DHALF, qkOut, qkIn, k1, DHALF, qkOut, qkIn,
        awL, SS, awOut, awIn, nullptr, mask, adjoin, DHEAD, HH, SS);
    gemm_wmma_bf16<3,0><<<gScore, 256, 0, stream>>>(
        q2, DHALF, qkOut, qkIn, k2, DHALF, qkOut, qkIn,
        awG, SS, awOut, awIn, nullptr, mask, dist, DHEAD, HH, SS);

    // ---- 4) Softmax in place on d_out + bf16 shadow for the AV GEMM ----
    softmax512<<<2 * BB * HH * SS, 128, 0, stream>>>(awL, awLb);

    // ---- 5) Transpose V per (b,h) to bf16: Vt[b,h][d,j] = V[b,j,h*64+d] ----
    dim3 gVt(DHEAD / 32, SS / 32, BB * HH);
    transpose_f32_bf16<<<gVt, tb, 0, stream>>>(
        v1, DHALF, (long)SS * DHALF, DHEAD,
        Vt1, SS, (long)HH * DHEAD * SS, (long)DHEAD * SS, SS, DHEAD, HH);
    transpose_f32_bf16<<<gVt, tb, 0, stream>>>(
        v2, DHALF, (long)SS * DHALF, DHEAD,
        Vt2, SS, (long)HH * DHEAD * SS, (long)DHEAD * SS, SS, DHEAD, HH);

    // ---- 6) ctx = aw @ V  (M=512, N=64, K=512, batched 128) -> bf16 ----
    dim3 gAV(SS / 128, DHEAD / 64, BB * HH);
    gemm_wmma_bf16<0,1><<<gAV, 256, 0, stream>>>(
        awLb, SS, awOut, awIn,
        Vt1, SS, (long)HH * DHEAD * SS, (long)DHEAD * SS,
        ctx1, DHALF, (long)SS * DHALF, DHEAD,
        nullptr, nullptr, nullptr, SS, HH, SS);
    gemm_wmma_bf16<0,1><<<gAV, 256, 0, stream>>>(
        awGb, SS, awOut, awIn,
        Vt2, SS, (long)HH * DHEAD * SS, (long)DHEAD * SS,
        ctx2, DHALF, (long)SS * DHALF, DHEAD,
        nullptr, nullptr, nullptr, SS, HH, SS);

    // ---- 7) Output projections -> concat halves of attn (fp32) ----
    gemm_wmma_bf16<0,0><<<gProj, 256, 0, stream>>>(
        ctx1, DHALF, 0, 0, Wt[3], DHALF, 0, 0,
        attn, DD, 0, 0, Bv[3], nullptr, nullptr, DHALF, 1, SS);
    gemm_wmma_bf16<0,0><<<gProj, 256, 0, stream>>>(
        ctx2, DHALF, 0, 0, Wt[7], DHALF, 0, 0,
        attn + DHALF, DD, 0, 0, Bv[7], nullptr, nullptr, DHALF, 1, SS);

    // ---- 8) out1 = LN(x + attn), fp32 + bf16 shadow ----
    add_ln1024<<<MM, 256, 0, stream>>>(x, attn, ln1g, ln1b, out1, out1b);

    // ---- 9) hidden = gelu(out1 @ w_ff1 + b_ff1) -> bf16 ----
    gemm_wmma_bf16<1,1><<<dim3(MM / 128, DFFN / 64, 1), 256, 0, stream>>>(
        out1b, DD, 0, 0, wff1t, DD, 0, 0,
        hidden, DFFN, 0, 0, bff1, nullptr, nullptr, DD, 1, SS);

    // ---- 10) ffnout = hidden @ w_ff2 + b_ff2 (fp32) ----
    gemm_wmma_bf16<0,0><<<dim3(MM / 128, DD / 64, 1), 256, 0, stream>>>(
        hidden, DFFN, 0, 0, wff2t, DFFN, 0, 0,
        ffnout, DD, 0, 0, bff2, nullptr, nullptr, DFFN, 1, SS);

    // ---- 11) out2 = LN(out1 + ffnout) -> d_out ----
    add_ln1024<<<MM, 256, 0, stream>>>(out1, ffnout, ln2g, ln2b, out, nullptr);
}
